// SlotAttention_9921374454103
// MI455X (gfx1250) — matrix operand order repaired and checked by reading, not compiled, not run on hardware
//
#include <hip/hip_runtime.h>
#include <math.h>

// ---------------------------------------------------------------------------
// Slot Attention forward for gfx1250 (MI455X), bf16 WMMA + fp32 accumulate.
// B=32, N=2048, D=512, NS=8, ITERS=3.  Workspace use ~220 MB.
// ---------------------------------------------------------------------------

typedef __bf16 bf16;
typedef __attribute__((ext_vector_type(16))) __bf16         bf16x16;
typedef __attribute__((ext_vector_type(8)))  float          f32x8;
typedef __attribute__((ext_vector_type(8)))  unsigned short u16x8;

#define B_    32
#define N_    2048
#define D_    512
#define NS_   8
#define MROWS (B_ * N_)    // 65536
#define SROWS (B_ * NS_)   // 256

union Frag {
    bf16x16 v;
    u16x8   u[2];
};

__device__ __forceinline__ f32x8 wmma_bf16(const Frag& a, const Frag& b, f32x8 c) {
    return __builtin_amdgcn_wmma_f32_16x16x32_bf16(
        /*neg_a=*/false, a.v, /*neg_b=*/false, b.v,
        /*c_mod=*/(short)0, c, /*reuse_a=*/false, /*reuse_b=*/false);
}

// ---------------------------------------------------------------------------
// Generic GEMM:  C[M,N] = act( A[M,K] @ W[N,K]^T + bias )
// A, W bf16 row-major (both K-contiguous).  One wave -> 16x64 output strip.
// Optional fp32 out (row-major), optional bf16 out (row-major or transposed
// [N,M] with packed 16B stores).  gelu = exact erf GELU.
// ---------------------------------------------------------------------------
__global__ void gemm_bf16_kernel(const bf16* __restrict__ A,
                                 const bf16* __restrict__ W,
                                 const float* __restrict__ bias,
                                 float* __restrict__ Cf,
                                 bf16*  __restrict__ Cb,
                                 int M, int N, int K,
                                 int gelu, int transpose_cb)
{
    const int tid  = blockIdx.x * blockDim.x + threadIdx.x;
    const int wid  = tid >> 5;
    const int lane = tid & 31;
    const int lo   = lane & 15;
    const int hi   = lane >> 4;

    const int nstrips = N >> 6;          // 64-wide strips
    const int mt = wid / nstrips;
    const int ns = wid % nstrips;
    const int m0 = mt << 4;
    const int n0 = ns << 6;

    f32x8 acc[4] = {f32x8{}, f32x8{}, f32x8{}, f32x8{}};

    const bf16* aptr  = A + (size_t)(m0 + lo) * K;
    const bf16* wbase = W + (size_t)(n0 + lo) * K;

    for (int k = 0; k < K; k += 32) {
        __builtin_prefetch(aptr + k + 256, 0, 0);   // global_prefetch_b8
        Frag a;
        a.u[0] = *(const u16x8*)(aptr + k + 8 * hi);
        a.u[1] = *(const u16x8*)(aptr + k + 16 + 8 * hi);
#pragma unroll
        for (int t = 0; t < 4; ++t) {
            Frag b;
            const bf16* wp = wbase + (size_t)t * 16 * K + k;
            b.u[0] = *(const u16x8*)(wp + 8 * hi);
            b.u[1] = *(const u16x8*)(wp + 16 + 8 * hi);
            acc[t] = wmma_bf16(a, b, acc[t]);
        }
    }

#pragma unroll
    for (int t = 0; t < 4; ++t) {
        const int col = n0 + t * 16 + lo;
        const float bv = bias ? bias[col] : 0.0f;
        float xv[8];
#pragma unroll
        for (int r = 0; r < 8; ++r) {
            float x = acc[t][r] + bv;
            if (gelu) x = 0.5f * x * (1.0f + erff(x * 0.70710678118654752f));
            xv[r] = x;
        }
        if (Cf) {
#pragma unroll
            for (int r = 0; r < 8; ++r)
                Cf[(size_t)(m0 + r + 8 * hi) * N + col] = xv[r];
        }
        if (Cb) {
            if (transpose_cb) {
                u16x8 pk;
#pragma unroll
                for (int r = 0; r < 8; ++r)
                    pk[r] = __builtin_bit_cast(unsigned short, (bf16)xv[r]);
                *(u16x8*)(Cb + (size_t)col * M + m0 + 8 * hi) = pk;
            } else {
#pragma unroll
                for (int r = 0; r < 8; ++r)
                    Cb[(size_t)(m0 + r + 8 * hi) * N + col] = (bf16)xv[r];
            }
        }
    }
}

// ---------------------------------------------------------------------------
// dots[b,i,j] = SCALE * sum_d q[b,i,d] * k[b,j,d]   (i padded 8->16 w/ zeros)
// One wave per (batch, 64-token strip).  K loop over D=512.
// ---------------------------------------------------------------------------
__global__ void dots_kernel(const bf16* __restrict__ Q,   // [B, NS, D]
                            const bf16* __restrict__ Kb,  // [B*N, D]
                            float* __restrict__ dots)     // [B, NS, N]
{
    const int tid  = blockIdx.x * blockDim.x + threadIdx.x;
    const int wid  = tid >> 5;
    const int lane = tid & 31;
    const int lo   = lane & 15;
    const int hi   = lane >> 4;

    const int b  = wid >> 5;       // 32 strips of 64 tokens per batch
    const int n0 = (wid & 31) << 6;

    f32x8 acc[4] = {f32x8{}, f32x8{}, f32x8{}, f32x8{}};
    const u16x8 zero8 = {};

    const bf16* aptr  = Q + ((size_t)b * NS_ + lo) * D_;        // valid lo<8
    const bf16* kbase = Kb + ((size_t)b * N_ + n0 + lo) * D_;

    for (int k = 0; k < D_; k += 32) {
        Frag a;
        if (lo < NS_) {
            a.u[0] = *(const u16x8*)(aptr + k + 8 * hi);
            a.u[1] = *(const u16x8*)(aptr + k + 16 + 8 * hi);
        } else {
            a.u[0] = zero8;
            a.u[1] = zero8;
        }
#pragma unroll
        for (int t = 0; t < 4; ++t) {
            Frag bb;
            const bf16* wp = kbase + (size_t)t * 16 * D_ + k;
            bb.u[0] = *(const u16x8*)(wp + 8 * hi);
            bb.u[1] = *(const u16x8*)(wp + 16 + 8 * hi);
            acc[t] = wmma_bf16(a, bb, acc[t]);
        }
    }

    if (hi == 0) {                 // rows 0..7 are the real slots
        const float scale = 0.04419417382415922f;  // 512^-0.5
#pragma unroll
        for (int t = 0; t < 4; ++t) {
            const int col = n0 + t * 16 + lo;
#pragma unroll
            for (int r = 0; r < 8; ++r)
                dots[((size_t)b * NS_ + r) * N_ + col] = acc[t][r] * scale;
        }
    }
}

// ---------------------------------------------------------------------------
// updates[b,i,d] = sum_j attn[b,i,j] * v[b,j,d]; v given transposed
// v_t[d, b*N+j] so B-frag loads are K(=j)-contiguous.
// ---------------------------------------------------------------------------
__global__ void updates_kernel(const bf16* __restrict__ Attn,  // [B, NS, N]
                               const bf16* __restrict__ Vt,    // [D, B*N]
                               float* __restrict__ Uf,         // [B, NS, D]
                               bf16*  __restrict__ Ub)         // [B, NS, D]
{
    const int tid  = blockIdx.x * blockDim.x + threadIdx.x;
    const int wid  = tid >> 5;
    const int lane = tid & 31;
    const int lo   = lane & 15;
    const int hi   = lane >> 4;

    const int b  = wid >> 3;       // 8 strips of 64 d-cols per batch
    const int n0 = (wid & 7) << 6;

    f32x8 acc[4] = {f32x8{}, f32x8{}, f32x8{}, f32x8{}};
    const u16x8 zero8 = {};

    const bf16* aptr  = Attn + ((size_t)b * NS_ + lo) * N_;
    const bf16* vbase = Vt + (size_t)(n0 + lo) * MROWS + (size_t)b * N_;

    for (int k = 0; k < N_; k += 32) {
        Frag a;
        if (lo < NS_) {
            a.u[0] = *(const u16x8*)(aptr + k + 8 * hi);
            a.u[1] = *(const u16x8*)(aptr + k + 16 + 8 * hi);
        } else {
            a.u[0] = zero8;
            a.u[1] = zero8;
        }
#pragma unroll
        for (int t = 0; t < 4; ++t) {
            Frag bb;
            const bf16* wp = vbase + (size_t)t * 16 * MROWS + k;
            bb.u[0] = *(const u16x8*)(wp + 8 * hi);
            bb.u[1] = *(const u16x8*)(wp + 16 + 8 * hi);
            acc[t] = wmma_bf16(a, bb, acc[t]);
        }
    }

    if (hi == 0) {
#pragma unroll
        for (int t = 0; t < 4; ++t) {
            const int col = n0 + t * 16 + lo;
#pragma unroll
            for (int r = 0; r < 8; ++r) {
                const size_t idx = ((size_t)b * NS_ + r) * D_ + col;
                Uf[idx] = acc[t][r];
                Ub[idx] = (bf16)acc[t][r];
            }
        }
    }
}

// ---------------------------------------------------------------------------
// Row LayerNorm (D=512) -> bf16 out.  One 256-thread block per row.
// ---------------------------------------------------------------------------
__global__ void ln_rows_kernel(const float* __restrict__ X,
                               const float* __restrict__ g,
                               const float* __restrict__ bb,
                               bf16* __restrict__ out)
{
    const int row = blockIdx.x;
    const int tid = threadIdx.x;
    const float* xr = X + (size_t)row * D_;

    float s = 0.f, sq = 0.f;
    for (int i = tid; i < D_; i += 256) {
        const float v = xr[i];
        s  += v;
        sq += v * v;
    }
#pragma unroll
    for (int off = 16; off > 0; off >>= 1) {
        s  += __shfl_xor(s,  off, 32);
        sq += __shfl_xor(sq, off, 32);
    }
    __shared__ float ss[8], ssq[8];
    const int w = tid >> 5;
    if ((tid & 31) == 0) { ss[w] = s; ssq[w] = sq; }
    __syncthreads();
    float S = 0.f, SQ = 0.f;
#pragma unroll
    for (int i = 0; i < 8; ++i) { S += ss[i]; SQ += ssq[i]; }

    const float mean = S * (1.0f / D_);
    const float var  = SQ * (1.0f / D_) - mean * mean;
    const float inv  = rsqrtf(var + 1e-5f);
    for (int i = tid; i < D_; i += 256)
        out[(size_t)row * D_ + i] = (bf16)((xr[i] - mean) * inv * g[i] + bb[i]);
}

// softmax over the slot axis (8 values, stride N) + EPS, in place.
__global__ void softmax_slots_kernel(float* __restrict__ dots)
{
    const int tid = blockIdx.x * blockDim.x + threadIdx.x;  // B*N threads
    const int b = tid >> 11;
    const int j = tid & (N_ - 1);
    float* p = dots + (size_t)b * NS_ * N_ + j;

    float v[NS_];
    float mx = -3.0e38f;
#pragma unroll
    for (int i = 0; i < NS_; ++i) { v[i] = p[(size_t)i * N_]; mx = fmaxf(mx, v[i]); }
    float sum = 0.f;
#pragma unroll
    for (int i = 0; i < NS_; ++i) { v[i] = __expf(v[i] - mx); sum += v[i]; }
    const float inv = 1.0f / sum;
#pragma unroll
    for (int i = 0; i < NS_; ++i) p[(size_t)i * N_] = v[i] * inv + 1e-8f;
}

// renormalize each (b,i) row over tokens, emit bf16 attn.
__global__ void renorm_kernel(const float* __restrict__ attn,
                              bf16* __restrict__ attn_bf)
{
    const int row = blockIdx.x;          // SROWS rows
    const int tid = threadIdx.x;
    const float* p = attn + (size_t)row * N_;

    float s = 0.f;
    for (int j = tid; j < N_; j += 256) s += p[j];
#pragma unroll
    for (int off = 16; off > 0; off >>= 1) s += __shfl_xor(s, off, 32);
    __shared__ float ss[8];
    if ((tid & 31) == 0) ss[tid >> 5] = s;
    __syncthreads();
    float S = 0.f;
#pragma unroll
    for (int i = 0; i < 8; ++i) S += ss[i];

    const float inv = 1.0f / S;
    for (int j = tid; j < N_; j += 256)
        attn_bf[(size_t)row * N_ + j] = (bf16)(p[j] * inv);
}

// GRUCell + residual: slots = u + [(1-z)*tanh(i_n + r*h_n) + z*h], in place.
__global__ void gru_kernel(const float* __restrict__ gi,
                           const float* __restrict__ gh,
                           const float* __restrict__ uf,
                           float* __restrict__ slots)
{
    const int tid = blockIdx.x * blockDim.x + threadIdx.x;  // SROWS*D
    const int row = tid >> 9;
    const int c   = tid & (D_ - 1);
    const float* gir = gi + (size_t)row * (3 * D_);
    const float* ghr = gh + (size_t)row * (3 * D_);

    const float r  = 1.0f / (1.0f + __expf(-(gir[c] + ghr[c])));
    const float z  = 1.0f / (1.0f + __expf(-(gir[D_ + c] + ghr[D_ + c])));
    const float nn = tanhf(gir[2 * D_ + c] + r * ghr[2 * D_ + c]);
    const float h  = slots[tid];
    slots[tid] = uf[tid] + (1.0f - z) * nn + z * h;
}

__global__ void init_slots_kernel(const float* __restrict__ noise,
                                  const float* __restrict__ mu,
                                  const float* __restrict__ ls,
                                  float* __restrict__ slots)
{
    const int tid = blockIdx.x * blockDim.x + threadIdx.x;  // SROWS*D
    const int d = tid & (D_ - 1);
    slots[tid] = mu[d] + __expf(ls[d]) * noise[tid];
}

__global__ void cvt_kernel(const float* __restrict__ in, bf16* __restrict__ out, int n)
{
    const int tid = blockIdx.x * blockDim.x + threadIdx.x;
    if (tid < n) out[tid] = (bf16)in[tid];
}

// ---------------------------------------------------------------------------
extern "C" void kernel_launch(void* const* d_in, const int* in_sizes, int n_in,
                              void* d_out, int out_size, void* d_ws, size_t ws_size,
                              hipStream_t stream)
{
    const float* inputs  = (const float*)d_in[0];
    const float* noise   = (const float*)d_in[1];
    const float* mu      = (const float*)d_in[2];
    const float* lsg     = (const float*)d_in[3];
    const float* k_w     = (const float*)d_in[4];
    const float* k_b     = (const float*)d_in[5];
    const float* v_w     = (const float*)d_in[6];
    const float* v_b     = (const float*)d_in[7];
    const float* w_ih    = (const float*)d_in[8];
    const float* w_hh    = (const float*)d_in[9];
    const float* b_ih    = (const float*)d_in[10];
    const float* b_hh    = (const float*)d_in[11];
    const float* ln_in_g = (const float*)d_in[12];
    const float* ln_in_b = (const float*)d_in[13];
    const float* ln_s_g  = (const float*)d_in[14];
    const float* ln_s_b  = (const float*)d_in[15];
    const float* ln_p_g  = (const float*)d_in[16];
    const float* ln_p_b  = (const float*)d_in[17];
    const float* m1_w    = (const float*)d_in[18];
    const float* m1_b    = (const float*)d_in[19];
    const float* m2_w    = (const float*)d_in[20];
    const float* m2_b    = (const float*)d_in[21];
    const float* m3_w    = (const float*)d_in[22];
    const float* m3_b    = (const float*)d_in[23];
    const float* m4_w    = (const float*)d_in[24];
    const float* m4_b    = (const float*)d_in[25];

    char* ws = (char*)d_ws;
    size_t off = 0;
    auto alloc = [&](size_t bytes) -> void* {
        void* p = ws + off;
        off = (off + bytes + 255) & ~(size_t)255;
        return p;
    };

    bf16* x_bf   = (bf16*)alloc((size_t)MROWS * D_ * 2);
    bf16* k_bf   = (bf16*)alloc((size_t)MROWS * D_ * 2);
    bf16* vt_bf  = (bf16*)alloc((size_t)MROWS * D_ * 2);   // transposed [D, B*N]
    bf16* wk_bf  = (bf16*)alloc((size_t)512 * 512 * 2);
    bf16* wv_bf  = (bf16*)alloc((size_t)512 * 512 * 2);
    bf16* wih_bf = (bf16*)alloc((size_t)1536 * 512 * 2);
    bf16* whh_bf = (bf16*)alloc((size_t)1536 * 512 * 2);
    bf16* w1_bf  = (bf16*)alloc((size_t)1024 * 512 * 2);
    bf16* w2_bf  = (bf16*)alloc((size_t)1024 * 1024 * 2);
    bf16* w3_bf  = (bf16*)alloc((size_t)512 * 1024 * 2);
    bf16* w4_bf  = (bf16*)alloc((size_t)256 * 512 * 2);
    float* slots_f  = (float*)alloc((size_t)SROWS * D_ * 4);
    bf16*  slots_bf = (bf16*) alloc((size_t)SROWS * D_ * 2);
    bf16*  q_bf     = (bf16*) alloc((size_t)SROWS * D_ * 2);
    float* dots_f   = (float*)alloc((size_t)SROWS * N_ * 4);
    bf16*  attn_bf  = (bf16*) alloc((size_t)SROWS * N_ * 2);
    float* u_f      = (float*)alloc((size_t)SROWS * D_ * 4);
    bf16*  u_bf     = (bf16*) alloc((size_t)SROWS * D_ * 2);
    float* gi_f     = (float*)alloc((size_t)SROWS * 1536 * 4);
    float* gh_f     = (float*)alloc((size_t)SROWS * 1536 * 4);
    bf16*  s_bf     = (bf16*) alloc((size_t)SROWS * D_ * 2);
    bf16*  a1_bf    = (bf16*) alloc((size_t)SROWS * 1024 * 2);
    bf16*  a2_bf    = (bf16*) alloc((size_t)SROWS * 1024 * 2);
    bf16*  a3_bf    = (bf16*) alloc((size_t)SROWS * 512 * 2);

    auto cvt = [&](const float* src, bf16* dst, int n) {
        cvt_kernel<<<(n + 255) / 256, 256, 0, stream>>>(src, dst, n);
    };
    auto gemm = [&](const bf16* A, const bf16* W, const float* bias,
                    float* Cf, bf16* Cb, int M, int N, int K, int gelu, int tcb) {
        const int waves = (M / 16) * (N / 64);
        gemm_bf16_kernel<<<(waves * 32) / 256, 256, 0, stream>>>(
            A, W, bias, Cf, Cb, M, N, K, gelu, tcb);
    };

    // Weight -> bf16
    cvt(k_w,  wk_bf,  512 * 512);
    cvt(v_w,  wv_bf,  512 * 512);
    cvt(w_ih, wih_bf, 1536 * 512);
    cvt(w_hh, whh_bf, 1536 * 512);
    cvt(m1_w, w1_bf,  1024 * 512);
    cvt(m2_w, w2_bf,  1024 * 1024);
    cvt(m3_w, w3_bf,  512 * 1024);
    cvt(m4_w, w4_bf,  256 * 512);

    // slots = mu + exp(logsigma) * noise
    init_slots_kernel<<<(SROWS * D_) / 256, 256, 0, stream>>>(noise, mu, lsg, slots_f);

    // x = LN(inputs) -> bf16 ; k = x@k_w^T+k_b ; v^T = (x@v_w^T+v_b)^T
    ln_rows_kernel<<<MROWS, 256, 0, stream>>>(inputs, ln_in_g, ln_in_b, x_bf);
    gemm(x_bf, wk_bf, k_b, nullptr, k_bf, MROWS, 512, 512, 0, 0);
    gemm(x_bf, wv_bf, v_b, nullptr, vt_bf, MROWS, 512, 512, 0, 1);

    float* upd_out = (float*)d_out;                 // [B, NS, D] = 131072 floats
    float* mlp_out = (float*)d_out + SROWS * D_;    // [B, NS, 256]

    for (int it = 0; it < 3; ++it) {
        ln_rows_kernel<<<SROWS, 256, 0, stream>>>(slots_f, ln_s_g, ln_s_b, q_bf);
        dots_kernel<<<(32 * 32 * 32) / 256, 256, 0, stream>>>(q_bf, k_bf, dots_f);
        softmax_slots_kernel<<<(B_ * N_) / 256, 256, 0, stream>>>(dots_f);
        renorm_kernel<<<SROWS, 256, 0, stream>>>(dots_f, attn_bf);

        float* uft = (it == 2) ? upd_out : u_f;     // final einsum is output[0]
        updates_kernel<<<(32 * 8 * 32) / 256, 256, 0, stream>>>(attn_bf, vt_bf, uft, u_bf);

        cvt(slots_f, slots_bf, SROWS * D_);         // slots_prev for gh
        gemm(u_bf,     wih_bf, b_ih, gi_f, nullptr, SROWS, 1536, 512, 0, 0);
        gemm(slots_bf, whh_bf, b_hh, gh_f, nullptr, SROWS, 1536, 512, 0, 0);
        gru_kernel<<<(SROWS * D_) / 256, 256, 0, stream>>>(gi_f, gh_f, uft, slots_f);
    }

    // Output MLP
    ln_rows_kernel<<<SROWS, 256, 0, stream>>>(slots_f, ln_p_g, ln_p_b, s_bf);
    gemm(s_bf,  w1_bf, m1_b, nullptr, a1_bf, SROWS, 1024, 512,  1, 0);
    gemm(a1_bf, w2_bf, m2_b, nullptr, a2_bf, SROWS, 1024, 1024, 1, 0);
    gemm(a2_bf, w3_bf, m3_b, nullptr, a3_bf, SROWS, 512,  1024, 1, 0);
    gemm(a3_bf, w4_bf, m4_b, mlp_out, nullptr, SROWS, 256, 512, 0, 0);

    (void)in_sizes; (void)n_in; (void)out_size; (void)ws_size;
}
